// GAT_5643587027020
// MI455X (gfx1250) — matrix operand order repaired and checked by reading, not compiled, not run on hardware
//
#include <hip/hip_runtime.h>
#include <hip/hip_fp16.h>

typedef __attribute__((ext_vector_type(16))) _Float16 v16h;
typedef __attribute__((ext_vector_type(8)))  _Float16 v8h;
typedef __attribute__((ext_vector_type(8)))  float    v8f;

union AFrag { v16h v; v8h h[2]; };
union Acc   { v8f  v; float f[8]; };

#define IN_CH 512
#define HIDX  512   // HEADS*HID
#define NEG_SLOPE 0.2f

// ---------------- conversion / swizzle ----------------
__global__ void k_cvt_x(const float* __restrict__ x, _Float16* __restrict__ xh, int n) {
    int t = blockIdx.x * 256 + threadIdx.x;
    if (t < n) xh[t] = (_Float16)x[t];
}

// W1 [512 K x 512 N] row-major -> fragment-major f16:
// frag = ntile*16 + ks ; within frag: lane*16 + i
// lane: col = ntile*16 + (lane&15) ; K = ks*32 + (lane>>4)*16 + i
__global__ void k_swz_w(const float* __restrict__ W1, _Float16* __restrict__ wh) {
    int d = blockIdx.x * 256 + threadIdx.x;           // 0 .. 512*512-1
    int frag = d >> 9, lane = (d >> 4) & 31, i = d & 15;
    int ntile = frag >> 4, ks = frag & 15;
    int col = (ntile << 4) + (lane & 15);
    int K   = (ks << 5) + ((lane >> 4) << 4) + i;
    wh[d] = (_Float16)W1[K * HIDX + col];
}

// ---------------- GEMM1 via WMMA: h = x @ W1 ----------------
// grid = N/16 blocks, 256 threads (8 waves). Wave w: rows m0..m0+15, cols w*64..w*64+63.
__global__ void __launch_bounds__(256) k_gemm1(const _Float16* __restrict__ xh,
                                               const _Float16* __restrict__ wh,
                                               float* __restrict__ h) {
    const int lane = threadIdx.x & 31;
    const int wave = threadIdx.x >> 5;
    const int m0   = blockIdx.x << 4;
    const int row  = m0 + (lane & 15);
    const int khi  = (lane >> 4) << 4;                // 0 or 16

    Acc acc[4];
#pragma unroll
    for (int s = 0; s < 4; ++s)
#pragma unroll
        for (int j = 0; j < 8; ++j) acc[s].f[j] = 0.f;

    const _Float16* xrow = xh + (size_t)row * IN_CH;
    for (int ks = 0; ks < 16; ++ks) {
        AFrag A;
        A.h[0] = *(const v8h*)(xrow + ks * 32 + khi);        // K = kq..kq+7
        A.h[1] = *(const v8h*)(xrow + ks * 32 + khi + 8);    // K = kq+8..kq+15
#pragma unroll
        for (int s = 0; s < 4; ++s) {
            const int ntile = (wave << 2) + s;
            const _Float16* bp = wh + ((size_t)((ntile << 4) + ks) << 9) + (lane << 4);
            AFrag B;
            B.h[0] = *(const v8h*)(bp);
            B.h[1] = *(const v8h*)(bp + 8);
            acc[s].v = __builtin_amdgcn_wmma_f32_16x16x32_f16(
                false, A.v, false, B.v, (short)0, acc[s].v, false, false);
        }
    }
    const int colb = (wave << 6) + (lane & 15);
    const int rb   = m0 + ((lane >> 4) << 3);
#pragma unroll
    for (int s = 0; s < 4; ++s) {
        float* hp = h + (size_t)rb * HIDX + colb + (s << 4);
#pragma unroll
        for (int i = 0; i < 8; ++i) hp[(size_t)i * HIDX] = acc[s].f[i];
    }
}

// ---------------- attention scores layer 1 (wave per node) ----------------
__global__ void __launch_bounds__(256) k_attn1(const float* __restrict__ h,
        const float* __restrict__ att_s, const float* __restrict__ att_d,
        float* __restrict__ a_s, float* __restrict__ a_d, int N) {
    const int lane = threadIdx.x & 31;
    const int node = (blockIdx.x << 3) + (threadIdx.x >> 5);
    if (node >= N) return;
    float ps[4] = {0, 0, 0, 0}, pd[4] = {0, 0, 0, 0};
    const float* hr = h + (size_t)node * HIDX;
#pragma unroll
    for (int i = 0; i < 16; ++i) {
        int j = (i << 5) + lane;                 // head = j/128 = i/4, uniform per i
        float v = hr[j];
        ps[i >> 2] += v * att_s[j];
        pd[i >> 2] += v * att_d[j];
    }
#pragma unroll
    for (int hh = 0; hh < 4; ++hh) {
        float s = ps[hh], d = pd[hh];
        for (int off = 16; off; off >>= 1) { s += __shfl_down(s, off); d += __shfl_down(d, off); }
        if (lane == 0) { a_s[node * 4 + hh] = s; a_d[node * 4 + hh] = d; }
    }
}

// ---------------- init ----------------
__global__ void k_init1(float* __restrict__ o1, float* __restrict__ m,
                        float* __restrict__ dn, int N) {
    int t = blockIdx.x * 256 + threadIdx.x;
    if (t < N * HIDX) o1[t] = 0.f;
    if (t < N * 4) { m[t] = __int_as_float(0xFF800000); dn[t] = 0.f; }
}
__global__ void k_init2(float* __restrict__ out, float* __restrict__ m,
                        float* __restrict__ dn, int N) {
    int t = blockIdx.x * 256 + threadIdx.x;
    if (t < N * 2) out[t] = 0.f;
    if (t < N) { m[t] = __int_as_float(0xFF800000); dn[t] = 0.f; }
}

// ---------------- edge softmax passes ----------------
__device__ inline void atomicMaxF(float* a, float v) {
    if (v >= 0.f) atomicMax((int*)a, __float_as_int(v));
    else          atomicMin((unsigned int*)a, __float_as_uint(v));
}

template<int H>
__global__ void k_edge_max(const int* __restrict__ ei, const float* __restrict__ a_s,
        const float* __restrict__ a_d, float* __restrict__ eb, float* __restrict__ m,
        int NE, int ET) {
    int t = blockIdx.x * 256 + threadIdx.x;
    if (t >= ET * H) return;
    int e = t / H, hh = t - e * H;
    int s = e < NE ? ei[e] : e - NE;
    int d = e < NE ? ei[NE + e] : e - NE;
    float v = a_s[s * H + hh] + a_d[d * H + hh];
    v = v > 0.f ? v : NEG_SLOPE * v;
    eb[t] = v;
    atomicMaxF(m + d * H + hh, v);
}

template<int H>
__global__ void k_edge_exp(const int* __restrict__ ei, float* __restrict__ eb,
        const float* __restrict__ m, float* __restrict__ dn, int NE, int ET) {
    int t = blockIdx.x * 256 + threadIdx.x;
    if (t >= ET * H) return;
    int e = t / H, hh = t - e * H;
    int d = e < NE ? ei[NE + e] : e - NE;
    float ex = __expf(eb[t] - m[d * H + hh]);
    eb[t] = ex;
    atomicAdd(dn + d * H + hh, ex);
}

// ---------------- aggregation ----------------
// one block (128 threads) per edge; 4 channels per thread
__global__ void __launch_bounds__(128) k_agg1(const int* __restrict__ ei,
        const float* __restrict__ eb, const float* __restrict__ dn,
        const float* __restrict__ h, float* __restrict__ o1, int NE) {
    int e = blockIdx.x;
    int s = e < NE ? ei[e] : e - NE;
    int d = e < NE ? ei[NE + e] : e - NE;
    int c  = threadIdx.x << 2;
    int hh = c >> 7;
    float alpha = eb[e * 4 + hh] / (dn[d * 4 + hh] + 1e-16f);
    const float4 hv = *(const float4*)(h + (size_t)s * HIDX + c);
    float* op = o1 + (size_t)d * HIDX + c;
    atomicAdd(op + 0, alpha * hv.x);
    atomicAdd(op + 1, alpha * hv.y);
    atomicAdd(op + 2, alpha * hv.z);
    atomicAdd(op + 3, alpha * hv.w);
}

__global__ void k_elu(float* __restrict__ o1, const float* __restrict__ b1, int tot) {
    int t = blockIdx.x * 256 + threadIdx.x;
    if (t >= tot) return;
    float v = o1[t] + b1[t & (HIDX - 1)];
    o1[t] = v > 0.f ? v : expm1f(v);
}

// ---------------- layer 2: GEMM (512->2) + attention scores ----------------
__global__ void __launch_bounds__(256) k_gemm2(const float* __restrict__ h2,
        const float* __restrict__ W2, const float* __restrict__ as2,
        const float* __restrict__ ad2, float* __restrict__ g,
        float* __restrict__ a_s, float* __restrict__ a_d, int N) {
    const int lane = threadIdx.x & 31;
    const int node = (blockIdx.x << 3) + (threadIdx.x >> 5);
    if (node >= N) return;
    float s0 = 0.f, s1 = 0.f;
    const float* hr = h2 + (size_t)node * HIDX;
#pragma unroll
    for (int i = 0; i < 16; ++i) {
        int j = (i << 5) + lane;
        float v = hr[j];
        s0 += v * W2[j * 2];
        s1 += v * W2[j * 2 + 1];
    }
    for (int off = 16; off; off >>= 1) { s0 += __shfl_down(s0, off); s1 += __shfl_down(s1, off); }
    if (lane == 0) {
        g[node * 2] = s0; g[node * 2 + 1] = s1;
        a_s[node] = s0 * as2[0] + s1 * as2[1];
        a_d[node] = s0 * ad2[0] + s1 * ad2[1];
    }
}

__global__ void k_agg2(const int* __restrict__ ei, const float* __restrict__ eb,
        const float* __restrict__ dn, const float* __restrict__ g,
        float* __restrict__ out, int NE, int ET) {
    int e = blockIdx.x * 256 + threadIdx.x;
    if (e >= ET) return;
    int s = e < NE ? ei[e] : e - NE;
    int d = e < NE ? ei[NE + e] : e - NE;
    float alpha = eb[e] / (dn[d] + 1e-16f);
    atomicAdd(out + d * 2,     alpha * g[s * 2]);
    atomicAdd(out + d * 2 + 1, alpha * g[s * 2 + 1]);
}

__global__ void k_bias2(float* __restrict__ out, const float* __restrict__ b2, int N2) {
    int t = blockIdx.x * 256 + threadIdx.x;
    if (t < N2) out[t] += b2[t & 1];
}

// ---------------- host ----------------
extern "C" void kernel_launch(void* const* d_in, const int* in_sizes, int n_in,
                              void* d_out, int out_size, void* d_ws, size_t ws_size,
                              hipStream_t stream) {
    const float* x   = (const float*)d_in[0];
    const int*   ei  = (const int*)  d_in[1];
    const float* W1  = (const float*)d_in[2];
    const float* as1 = (const float*)d_in[3];
    const float* ad1 = (const float*)d_in[4];
    const float* b1  = (const float*)d_in[5];
    const float* W2  = (const float*)d_in[6];
    const float* as2 = (const float*)d_in[7];
    const float* ad2 = (const float*)d_in[8];
    const float* b2  = (const float*)d_in[9];
    float* out = (float*)d_out;

    const int N  = in_sizes[0] / IN_CH;     // 10000
    const int NE = in_sizes[1] / 2;         // 320000
    const int ET = NE + N;                  // +self loops

    char* p = (char*)d_ws;
    auto alloc = [&](size_t bytes) { void* r = (void*)p; p += (bytes + 255) & ~(size_t)255; return r; };
    _Float16* xh  = (_Float16*)alloc((size_t)N * IN_CH * 2);
    _Float16* wh  = (_Float16*)alloc((size_t)IN_CH * HIDX * 2);
    float* h      = (float*)alloc((size_t)N * HIDX * 4);
    float* o1     = (float*)alloc((size_t)N * HIDX * 4);
    float* a_s1   = (float*)alloc((size_t)N * 4 * 4);
    float* a_d1   = (float*)alloc((size_t)N * 4 * 4);
    float* m1     = (float*)alloc((size_t)N * 4 * 4);
    float* dn1    = (float*)alloc((size_t)N * 4 * 4);
    float* e1     = (float*)alloc((size_t)ET * 4 * 4);
    float* g      = (float*)alloc((size_t)N * 2 * 4);
    float* a_s2   = (float*)alloc((size_t)N * 4);
    float* a_d2   = (float*)alloc((size_t)N * 4);
    float* m2     = (float*)alloc((size_t)N * 4);
    float* dn2    = (float*)alloc((size_t)N * 4);
    float* e2     = (float*)alloc((size_t)ET * 4);

    const int tx = N * IN_CH;
    k_cvt_x<<<(tx + 255) / 256, 256, 0, stream>>>(x, xh, tx);
    k_swz_w<<<(IN_CH * HIDX) / 256, 256, 0, stream>>>(W1, wh);
    k_gemm1<<<N / 16, 256, 0, stream>>>(xh, wh, h);
    k_attn1<<<(N + 7) / 8, 256, 0, stream>>>(h, as1, ad1, a_s1, a_d1, N);
    k_init1<<<(N * HIDX + 255) / 256, 256, 0, stream>>>(o1, m1, dn1, N);
    k_edge_max<4><<<(ET * 4 + 255) / 256, 256, 0, stream>>>(ei, a_s1, a_d1, e1, m1, NE, ET);
    k_edge_exp<4><<<(ET * 4 + 255) / 256, 256, 0, stream>>>(ei, e1, m1, dn1, NE, ET);
    k_agg1<<<ET, 128, 0, stream>>>(ei, e1, dn1, h, o1, NE);
    k_elu<<<(N * HIDX + 255) / 256, 256, 0, stream>>>(o1, b1, N * HIDX);
    k_gemm2<<<(N + 7) / 8, 256, 0, stream>>>(o1, W2, as2, ad2, g, a_s2, a_d2, N);
    k_init2<<<(N * 2 + 255) / 256, 256, 0, stream>>>(out, m2, dn2, N);
    k_edge_max<1><<<(ET + 255) / 256, 256, 0, stream>>>(ei, a_s2, a_d2, e2, m2, NE, ET);
    k_edge_exp<1><<<(ET + 255) / 256, 256, 0, stream>>>(ei, e2, m2, dn2, NE, ET);
    k_agg2<<<(ET + 255) / 256, 256, 0, stream>>>(ei, e2, dn2, g, out, NE, ET);
    k_bias2<<<(N * 2 + 255) / 256, 256, 0, stream>>>(out, b2, N * 2);
}
